// GraphAttentionLayer_53772990545958
// MI455X (gfx1250) — matrix operand order repaired and checked by reading, not compiled, not run on hardware
//
#include <hip/hip_runtime.h>
#include <hip/hip_bf16.h>

// ---------------------------------------------------------------------------
// GraphAttentionLayer for MI455X (gfx1250, wave32, WMMA bf16 -> f32 accum)
// B=8, N=1024, D=512, H=8, DK=64
// ---------------------------------------------------------------------------

#define Bsz 8
#define Nseq 1024
#define Dmod 512
#define Hn 8
#define DKd 64
#define MROWS (Bsz * Nseq)          // 8192

typedef __attribute__((ext_vector_type(16))) __bf16 v16bf;
typedef __attribute__((ext_vector_type(8)))  __bf16 v8bf;
typedef __attribute__((ext_vector_type(8)))  float  v8f;

__device__ __forceinline__ v8f wmma_bf16(v16bf a, v16bf b, v8f c) {
    return __builtin_amdgcn_wmma_f32_16x16x32_bf16(
        /*neg_a=*/false, a, /*neg_b=*/false, b,
        /*c_mod=*/(short)0, c, /*reuse_a=*/false, /*reuse_b=*/false);
}

// A-fragment (16x32 bf16, M = lane%16, K interleave per ISA 7.12.2):
// elem e<8  -> K = e + 8*half ; elem e>=8 -> K = 16 + (e-8) + 8*half
__device__ __forceinline__ v16bf load_a_frag(const __bf16* p, int ld, int lane) {
    int row = lane & 15, half = lane >> 4;
    const __bf16* q = p + (size_t)row * ld + half * 8;
    v8bf lo = *(const v8bf*)(q);
    v8bf hi = *(const v8bf*)(q + 16);
    v16bf r;
#pragma unroll
    for (int i = 0; i < 8; ++i) { r[i] = lo[i]; r[8 + i] = hi[i]; }
    return r;
}

// B-fragment (32x16 bf16): col N = lane%16, K interleave same as A-fragment
// (source rows hold the contraction dim contiguously).
__device__ __forceinline__ v16bf load_b_frag(const __bf16* p, int ld, int lane) {
    int col = lane & 15, half = lane >> 4;
    const __bf16* q = p + (size_t)col * ld + half * 8;
    v8bf lo = *(const v8bf*)(q);
    v8bf hi = *(const v8bf*)(q + 16);
    v16bf r;
#pragma unroll
    for (int i = 0; i < 8; ++i) { r[i] = lo[i]; r[8 + i] = hi[i]; }
    return r;
}

// ---------------------------------------------------------------------------
// Kernel 1: fp32 -> bf16 conversion of x and weights
// ---------------------------------------------------------------------------
__global__ void cvt_kernel(const float* __restrict__ x,
                           const float* __restrict__ wq, const float* __restrict__ wk,
                           const float* __restrict__ wv, const float* __restrict__ wo,
                           __bf16* xb, __bf16* wqb, __bf16* wkb, __bf16* wvb, __bf16* wob) {
    size_t i = (size_t)blockIdx.x * blockDim.x + threadIdx.x;
    const size_t NX = (size_t)MROWS * Dmod;      // 4194304
    const size_t NW = (size_t)Dmod * Dmod;       // 262144
    if (i < NX) xb[i] = (__bf16)x[i];
    if (i < NW) {
        wqb[i] = (__bf16)wq[i];
        wkb[i] = (__bf16)wk[i];
        wvb[i] = (__bf16)wv[i];
        wob[i] = (__bf16)wo[i];
    }
}

// ---------------------------------------------------------------------------
// Kernel 2: fused QKV projection.  One wave -> one 16x16 output tile.
// grid (512, 12), block 256 (8 waves). ntile 0..95 covers [Wq|Wk|Wv].
// Q,K stored [B,H,N,64]; V stored transposed [B,H,64,N].
// ---------------------------------------------------------------------------
__global__ __launch_bounds__(256) void qkv_kernel(
        const __bf16* __restrict__ xb,
        const __bf16* __restrict__ wqb, const __bf16* __restrict__ wkb,
        const __bf16* __restrict__ wvb,
        const float* __restrict__ bq, const float* __restrict__ bk,
        const float* __restrict__ bv,
        __bf16* __restrict__ qs, __bf16* __restrict__ ks, __bf16* __restrict__ vt) {
    int lane = threadIdx.x & 31;
    int wid  = threadIdx.x >> 5;
    int mtile = blockIdx.x;                       // 0..511
    int nt = blockIdx.y * 8 + wid;                // 0..95
    int proj = nt >> 5;                           // 0=q 1=k 2=v
    int ct = nt & 31;                             // column tile within proj
    const __bf16* w    = (proj == 0) ? wqb : (proj == 1) ? wkb : wvb;
    const float*  bias = (proj == 0) ? bq  : (proj == 1) ? bk  : bv;
    const __bf16* arow = xb + (size_t)mtile * 16 * Dmod;
    const __bf16* wrow = w  + (size_t)ct * 16 * Dmod;

    v8f acc = {};
    for (int k0 = 0; k0 < Dmod; k0 += 32) {
        v16bf a  = load_a_frag(arow + k0, Dmod, lane);
        v16bf bf = load_b_frag(wrow + k0, Dmod, lane);
        acc = wmma_bf16(a, bf, acc);
    }

    int ln = lane & 15, half = lane >> 4;
    int col = ct * 16 + ln;                       // output dim within proj
    float bias_v = bias[col];
    int h = col >> 6, dk = col & 63;
#pragma unroll
    for (int r = 0; r < 8; ++r) {
        int row = mtile * 16 + r + 8 * half;
        int b = row >> 10, n = row & (Nseq - 1);
        __bf16 o = (__bf16)(acc[r] + bias_v);
        size_t bh = (size_t)(b * Hn + h);
        if (proj == 0)      qs[(bh * Nseq + n) * DKd + dk] = o;
        else if (proj == 1) ks[(bh * Nseq + n) * DKd + dk] = o;
        else                vt[(bh * DKd + dk) * Nseq + n] = o;
    }
}

// ---------------------------------------------------------------------------
// Kernel 3: flash-style attention. One wave per (b,h,16-query tile).
// grid 512, block 256 (8 waves). Streams 32-key chunks; online softmax.
// ---------------------------------------------------------------------------
__global__ __launch_bounds__(256) void attn_kernel(
        const __bf16* __restrict__ qs, const __bf16* __restrict__ ks,
        const __bf16* __restrict__ vt, const float* __restrict__ adj,
        __bf16* __restrict__ ab) {
    __shared__ __align__(16) __bf16 plds[8][16 * 32];   // per-wave P tile
    int lane = threadIdx.x & 31;
    int wid  = threadIdx.x >> 5;
    int gw = blockIdx.x * 8 + wid;                // 0..4095
    int qt = gw & 63;
    int h  = (gw >> 6) & 7;
    int b  = gw >> 9;
    size_t bh = (size_t)(b * Hn + h);
    int ln = lane & 15, half = lane >> 4;

    const __bf16* qbase = qs + (bh * Nseq + (size_t)qt * 16) * DKd;
    const __bf16* kbase = ks + bh * Nseq * DKd;
    const __bf16* vbase = vt + bh * DKd * Nseq;
    const float*  adjq  = adj + ((size_t)b * Nseq + (size_t)qt * 16) * Nseq;

    v16bf qa0 = load_a_frag(qbase,      DKd, lane);   // dk 0..31
    v16bf qa1 = load_a_frag(qbase + 32, DKd, lane);   // dk 32..63

    float m[8], l[8];
    v8f o0 = {}, o1 = {}, o2 = {}, o3 = {};
#pragma unroll
    for (int r = 0; r < 8; ++r) { m[r] = -1e30f; l[r] = 0.0f; }

    for (int kc = 0; kc < 32; ++kc) {
        int key0 = kc * 32;
        const __bf16* kt0 = kbase + (size_t)key0 * DKd;
        const __bf16* kt1 = kt0 + 16 * DKd;
        v8f s0 = {}, s1 = {};
        s0 = wmma_bf16(qa0, load_b_frag(kt0,      DKd, lane), s0);
        s0 = wmma_bf16(qa1, load_b_frag(kt0 + 32, DKd, lane), s0);
        s1 = wmma_bf16(qa0, load_b_frag(kt1,      DKd, lane), s1);
        s1 = wmma_bf16(qa1, load_b_frag(kt1 + 32, DKd, lane), s1);

#pragma unroll
        for (int r = 0; r < 8; ++r) {
            int qrow = r + 8 * half;
            const float* arow = adjq + (size_t)qrow * Nseq + key0;
            float t0 = s0[r] * 0.125f + __logf(fmaxf(arow[ln],      1e-8f));
            float t1 = s1[r] * 0.125f + __logf(fmaxf(arow[16 + ln], 1e-8f));
            // row max across the 16-lane group (wave32: xor<=8 stays in group)
            float mx = fmaxf(t0, t1);
            mx = fmaxf(mx, __shfl_xor(mx, 1));
            mx = fmaxf(mx, __shfl_xor(mx, 2));
            mx = fmaxf(mx, __shfl_xor(mx, 4));
            mx = fmaxf(mx, __shfl_xor(mx, 8));
            float mn = fmaxf(m[r], mx);
            float alpha = __expf(m[r] - mn);
            m[r] = mn;
            l[r] *= alpha;
            o0[r] *= alpha; o1[r] *= alpha; o2[r] *= alpha; o3[r] *= alpha;
            float p0 = __expf(t0 - mn), p1 = __expf(t1 - mn);
            float rs = p0 + p1;
            rs += __shfl_xor(rs, 1);
            rs += __shfl_xor(rs, 2);
            rs += __shfl_xor(rs, 4);
            rs += __shfl_xor(rs, 8);
            l[r] += rs;
            plds[wid][qrow * 32 + ln]      = (__bf16)p0;
            plds[wid][qrow * 32 + 16 + ln] = (__bf16)p1;
        }
        asm volatile("" ::: "memory");             // keep ds stores before loads
        v16bf pf = load_a_frag(&plds[wid][0], 32, lane);
        asm volatile("" ::: "memory");

        const __bf16* vb0 = vbase + key0;          // element (key,dk)=vt[dk*N+key]
        o0 = wmma_bf16(pf, load_b_frag(vb0,              Nseq, lane), o0);
        o1 = wmma_bf16(pf, load_b_frag(vb0 + 16 * Nseq,  Nseq, lane), o1);
        o2 = wmma_bf16(pf, load_b_frag(vb0 + 32 * Nseq,  Nseq, lane), o2);
        o3 = wmma_bf16(pf, load_b_frag(vb0 + 48 * Nseq,  Nseq, lane), o3);
    }

#pragma unroll
    for (int r = 0; r < 8; ++r) {
        float inv = 1.0f / l[r];
        int qrow = qt * 16 + r + 8 * half;
        size_t base = ((size_t)b * Nseq + qrow) * Dmod + (size_t)h * DKd;
        ab[base + ln]      = (__bf16)(o0[r] * inv);
        ab[base + 16 + ln] = (__bf16)(o1[r] * inv);
        ab[base + 32 + ln] = (__bf16)(o2[r] * inv);
        ab[base + 48 + ln] = (__bf16)(o3[r] * inv);
    }
}

// ---------------------------------------------------------------------------
// Kernel 4: output projection + bias + residual. grid (512,4), block 256.
// ---------------------------------------------------------------------------
__global__ __launch_bounds__(256) void oproj_kernel(
        const __bf16* __restrict__ ab, const __bf16* __restrict__ wob,
        const float* __restrict__ bo, const float* __restrict__ x,
        float* __restrict__ tmp) {
    int lane = threadIdx.x & 31;
    int wid  = threadIdx.x >> 5;
    int mtile = blockIdx.x;
    int ct = blockIdx.y * 8 + wid;                // 0..31
    const __bf16* arow = ab  + (size_t)mtile * 16 * Dmod;
    const __bf16* wrow = wob + (size_t)ct * 16 * Dmod;

    v8f acc = {};
    for (int k0 = 0; k0 < Dmod; k0 += 32)
        acc = wmma_bf16(load_a_frag(arow + k0, Dmod, lane),
                        load_b_frag(wrow + k0, Dmod, lane), acc);

    int ln = lane & 15, half = lane >> 4;
    int col = ct * 16 + ln;
    float bb = bo[col];
#pragma unroll
    for (int r = 0; r < 8; ++r) {
        size_t idx = ((size_t)(mtile * 16 + r + 8 * half)) * Dmod + col;
        tmp[idx] = acc[r] + bb + x[idx];
    }
}

// ---------------------------------------------------------------------------
// Kernel 5: per-row LayerNorm. grid 8192, block 128.
// ---------------------------------------------------------------------------
__global__ __launch_bounds__(128) void ln_kernel(
        const float* __restrict__ tmp, const float* __restrict__ gamma,
        const float* __restrict__ beta, float* __restrict__ out) {
    int row = blockIdx.x;
    const float* p = tmp + (size_t)row * Dmod;
    float v[4], s = 0.f, s2 = 0.f;
#pragma unroll
    for (int i = 0; i < 4; ++i) {
        v[i] = p[threadIdx.x + i * 128];
        s += v[i]; s2 += v[i] * v[i];
    }
#pragma unroll
    for (int off = 16; off >= 1; off >>= 1) {
        s  += __shfl_xor(s,  off);
        s2 += __shfl_xor(s2, off);
    }
    __shared__ float ss[4], ss2[4];
    int w = threadIdx.x >> 5, lane = threadIdx.x & 31;
    if (lane == 0) { ss[w] = s; ss2[w] = s2; }
    __syncthreads();
    s  = ss[0]  + ss[1]  + ss[2]  + ss[3];
    s2 = ss2[0] + ss2[1] + ss2[2] + ss2[3];
    float mu  = s  * (1.0f / Dmod);
    float var = s2 * (1.0f / Dmod) - mu * mu;
    float rinv = rsqrtf(var + 1e-5f);
    float* o = out + (size_t)row * Dmod;
#pragma unroll
    for (int i = 0; i < 4; ++i) {
        int c = threadIdx.x + i * 128;
        o[c] = (v[i] - mu) * rinv * gamma[c] + beta[c];
    }
}

// ---------------------------------------------------------------------------
// Host launcher
// ---------------------------------------------------------------------------
extern "C" void kernel_launch(void* const* d_in, const int* in_sizes, int n_in,
                              void* d_out, int out_size, void* d_ws, size_t ws_size,
                              hipStream_t stream) {
    (void)in_sizes; (void)n_in; (void)out_size; (void)ws_size;
    const float* x     = (const float*)d_in[0];
    const float* adj   = (const float*)d_in[1];
    const float* Wq    = (const float*)d_in[2];
    const float* bq    = (const float*)d_in[3];
    const float* Wk    = (const float*)d_in[4];
    const float* bk    = (const float*)d_in[5];
    const float* Wv    = (const float*)d_in[6];
    const float* bv    = (const float*)d_in[7];
    const float* Wo    = (const float*)d_in[8];
    const float* bo    = (const float*)d_in[9];
    const float* gamma = (const float*)d_in[10];
    const float* beta  = (const float*)d_in[11];
    float* out = (float*)d_out;

    char* ws = (char*)d_ws;
    const size_t SZ_X  = (size_t)MROWS * Dmod * sizeof(__bf16);  // 8 MB
    const size_t SZ_W  = (size_t)Dmod * Dmod * sizeof(__bf16);   // 512 KB
    __bf16* xb  = (__bf16*)(ws);
    __bf16* wqb = (__bf16*)(ws + SZ_X);
    __bf16* wkb = (__bf16*)(ws + SZ_X + 1 * SZ_W);
    __bf16* wvb = (__bf16*)(ws + SZ_X + 2 * SZ_W);
    __bf16* wob = (__bf16*)(ws + SZ_X + 3 * SZ_W);
    char* ws2 = ws + SZ_X + 4 * SZ_W;
    __bf16* qsb  = (__bf16*)(ws2);
    __bf16* ksb  = (__bf16*)(ws2 + SZ_X);
    __bf16* vtb  = (__bf16*)(ws2 + 2 * SZ_X);
    __bf16* attb = (__bf16*)(ws2 + 3 * SZ_X);
    float*  tmpf = (float*) (ws2 + 4 * SZ_X);                    // 16 MB

    // 1) convert to bf16
    cvt_kernel<<<(MROWS * Dmod) / 256, 256, 0, stream>>>(
        x, Wq, Wk, Wv, Wo, xb, wqb, wkb, wvb, wob);

    // 2) QKV projection (WMMA)
    qkv_kernel<<<dim3(MROWS / 16, 12), 256, 0, stream>>>(
        xb, wqb, wkb, wvb, bq, bk, bv, qsb, ksb, vtb);

    // 3) attention (WMMA + online softmax, streamed over keys)
    attn_kernel<<<(Bsz * Hn * (Nseq / 16)) / 8, 256, 0, stream>>>(
        qsb, ksb, vtb, adj, attb);

    // 4) output projection + residual (WMMA)
    oproj_kernel<<<dim3(MROWS / 16, 4), 256, 0, stream>>>(
        attb, wob, bo, x, tmpf);

    // 5) LayerNorm
    ln_kernel<<<MROWS, 128, 0, stream>>>(tmpf, gamma, beta, out);
}
